// ResDGFLayer_32555852103875
// MI455X (gfx1250) — compile-verified, matrix operand order checked by reading, not run on hardware
//
#include <hip/hip_runtime.h>
#include <hip/hip_bf16.h>
#include <hip/hip_fp16.h>

// Problem constants (from setup_inputs)
#define B_ 8
#define N_ 2048
#define D_ 256
#define H_ 4

typedef __attribute__((ext_vector_type(16))) _Float16 v16h;
typedef __attribute__((ext_vector_type(8)))  float    v8f;

union V16H { uint4 q[2]; v16h v; _Float16 h[16]; };
union V8H  { uint4 q;    _Float16 h[8]; };

static __device__ __forceinline__ v8f wmma_f16(v16h a, v16h b, v8f c) {
  // D = A(16x32,f16) * B(32x16,f16) + C(16x16,f32)
  return __builtin_amdgcn_wmma_f32_16x16x32_f16(
      /*neg_a=*/false, a, /*neg_b=*/false, b,
      /*c_mod=*/(short)0, c, /*reuse_a=*/false, /*reuse_b=*/false);
}

// Issue this thread's 4 async 16B DMA transfers of one 32x256-half key tile
// into LDS (GVS mode: 64-bit SGPR base + 32-bit per-lane byte offset).
// Tracked on ASYNCcnt; completion enforced with s_wait_asynccnt.
static __device__ __forceinline__ void stage_tile_async(
    unsigned long long gbase, int keybase, _Float16 (*buf)[D_], int t) {
  #pragma unroll
  for (int i = 0; i < 4; ++i) {
    int cc  = i * 256 + t;                 // 1024 chunks of 8 halfs
    int key = cc >> 5;
    int d0  = (cc & 31) * 8;
    unsigned voff = (unsigned)(((keybase + key) * D_ + d0) * 2); // byte offset
    unsigned lds  = (unsigned)(uintptr_t)(&buf[key][d0]);        // LDS byte addr
    asm volatile("global_load_async_to_lds_b128 %0, %1, %2"
                 :: "v"(lds), "v"(voff), "s"(gbase) : "memory");
  }
}

// ---------------------------------------------------------------------------
// Kernel 1a: convert x -> f16, compute row squared norms
// ---------------------------------------------------------------------------
__launch_bounds__(256)
__global__ void prep_x_kernel(const float* __restrict__ x,
                              _Float16* __restrict__ xh,
                              float* __restrict__ sq) {
  const int wave = threadIdx.x >> 5, lane = threadIdx.x & 31;
  const int row  = blockIdx.x * 8 + wave;          // 0 .. B*N-1
  const float* src = x + (size_t)row * D_ + lane * 8;
  float4 f0 = *(const float4*)(src);
  float4 f1 = *(const float4*)(src + 4);
  V8H u;
  u.h[0] = (_Float16)f0.x; u.h[1] = (_Float16)f0.y;
  u.h[2] = (_Float16)f0.z; u.h[3] = (_Float16)f0.w;
  u.h[4] = (_Float16)f1.x; u.h[5] = (_Float16)f1.y;
  u.h[6] = (_Float16)f1.z; u.h[7] = (_Float16)f1.w;
  *(uint4*)(xh + (size_t)row * D_ + lane * 8) = u.q;
  float s = f0.x*f0.x + f0.y*f0.y + f0.z*f0.z + f0.w*f0.w
          + f1.x*f1.x + f1.y*f1.y + f1.z*f1.z + f1.w*f1.w;
  #pragma unroll
  for (int off = 16; off > 0; off >>= 1) s += __shfl_down(s, off);
  if (lane == 0) sq[row] = s;
}

// ---------------------------------------------------------------------------
// Kernel 1b: convert W -> f16
// ---------------------------------------------------------------------------
__launch_bounds__(256)
__global__ void prep_w_kernel(const float* __restrict__ W,
                              _Float16* __restrict__ wh) {
  const int i = (blockIdx.x * 256 + threadIdx.x) * 4;
  float4 f = *(const float4*)(W + i);
  union { unsigned long long q; _Float16 h[4]; } u;
  u.h[0] = (_Float16)f.x; u.h[1] = (_Float16)f.y;
  u.h[2] = (_Float16)f.z; u.h[3] = (_Float16)f.w;
  *(unsigned long long*)(wh + i) = u.q;
}

// ---------------------------------------------------------------------------
// Kernel 2: fused  S = Xr*Xk^T  ->  adj = mean_h exp(-relu(d)/denom_h)
//                 ->  y += adj * Xk     (never materializes (N,N) in HBM)
// Block = 64 rows of one batch, 8 waves: wave(wr 0..3, wc 0..1).
// Key tiles (32x256 f16) are double-buffered in LDS and filled by
// global_load_async_to_lds_b128 DMA overlapped with the WMMA/exp compute.
// ---------------------------------------------------------------------------
__launch_bounds__(256)
__global__ void adj_agg_kernel(const _Float16* __restrict__ xh,
                               const float* __restrict__ sq,
                               const float* __restrict__ log_sigmas,
                               _Float16* __restrict__ yh) {
  __shared__ __attribute__((aligned(16))) _Float16 skey [2][32][D_]; // 2x16KB
  __shared__ __attribute__((aligned(16))) _Float16 skeyT[D_][32];    // 16KB
  __shared__ __attribute__((aligned(16))) _Float16 adjBuf[4][16][32];// 4KB
  __shared__ float sqk[32];

  const int t    = threadIdx.x;
  const int wave = t >> 5, lane = t & 31;
  const int wr   = wave >> 1, wc = wave & 1;
  const int m    = lane & 15;            // M (or N) index inside 16x16 tile
  const int kb   = (lane >> 4) * 8;      // K sub-chunk base per WMMA layout
  const int rhalf = (lane >> 4) * 8;     // C-layout row offset

  const int bb      = blockIdx.x >> 5;               // batch
  const int rowbase = (blockIdx.x & 31) * 64 + wr * 16;
  const size_t batchOff = (size_t)bb * N_;
  const unsigned long long gbase =
      (unsigned long long)(uintptr_t)(xh + batchOff * (size_t)D_);

  // per-head 1/denom
  float inv[H_];
  #pragma unroll
  for (int h = 0; h < H_; ++h) {
    float sg = __expf(log_sigmas[h]);
    inv[h] = 1.0f / (2.0f * sg * sg + 1e-6f);
  }

  // Preload this wave's 16 rows (full K=256) as WMMA A operands, plus sq rows
  v16h A[8];
  {
    const _Float16* rp = xh + (batchOff + rowbase + m) * (size_t)D_;
    #pragma unroll
    for (int s = 0; s < 8; ++s) {
      V16H u;
      u.q[0] = *(const uint4*)(rp + s * 32 + kb);
      u.q[1] = *(const uint4*)(rp + s * 32 + kb + 16);
      A[s] = u.v;
    }
  }
  float sqr[8];
  #pragma unroll
  for (int r = 0; r < 8; ++r)
    sqr[r] = sq[batchOff + rowbase + r + rhalf];

  v8f zero = {};
  v8f acc[8];
  #pragma unroll
  for (int i = 0; i < 8; ++i) acc[i] = zero;

  const int nTiles = N_ / 32;  // 64

  // prologue: DMA tile 0 into buffer 0
  stage_tile_async(gbase, 0, skey[0], t);

  for (int kt = 0; kt < nTiles; ++kt) {
    const int cur = kt & 1;
    // Issue next tile's DMA first, then wait for the current tile: async
    // loads complete in order, so ASYNCcnt <= 4 means tile kt has landed
    // while tile kt+1 streams in behind the compute below.
    if (kt + 1 < nTiles) {
      stage_tile_async(gbase, (kt + 1) * 32, skey[cur ^ 1], t);
      asm volatile("s_wait_asynccnt 4" ::: "memory");
    } else {
      asm volatile("s_wait_asynccnt 0" ::: "memory");
    }
    __syncthreads();   // all waves' DMA for tile kt visible in LDS

    // ---- build transposed copy (B operand of aggregation GEMM) ----
    const int keybase = kt * 32;
    #pragma unroll
    for (int i = 0; i < 4; ++i) {
      int cc  = i * 256 + t;
      int key = cc >> 5;
      int d0  = (cc & 31) * 8;
      V8H u; u.q = *(const uint4*)&skey[cur][key][d0];
      #pragma unroll
      for (int j = 0; j < 8; ++j) skeyT[d0 + j][key] = u.h[j];
    }
    if (t < 32) sqk[t] = sq[batchOff + keybase + t];
    __syncthreads();

    // ---- S(16x16) = Xrow(16x256) * Xkey^T, this wave's 16-key half ----
    v8f S = zero;
    {
      const _Float16* bp = &skey[cur][wc * 16 + m][0]; // B[k][n]=xkey[n][k]
      #pragma unroll
      for (int s = 0; s < 8; ++s) {
        V16H u;
        u.q[0] = *(const uint4*)(bp + s * 32 + kb);
        u.q[1] = *(const uint4*)(bp + s * 32 + kb + 16);
        S = wmma_f16(A[s], u.v, S);
      }
    }
    // ---- adjacency: mean over heads of exp(-relu(d)/denom_h) ----
    #pragma unroll
    for (int r = 0; r < 8; ++r) {
      float d = sqr[r] + sqk[wc * 16 + m] - 2.0f * S[r];
      d = fmaxf(d, 0.0f);
      float a = 0.25f * (__expf(-d * inv[0]) + __expf(-d * inv[1]) +
                         __expf(-d * inv[2]) + __expf(-d * inv[3]));
      adjBuf[wr][r + rhalf][wc * 16 + m] = (_Float16)a;
    }
    __syncthreads();  // adjBuf complete across both wc halves

    // ---- acc(16x128) += adj(16x32) * Xkey(32x128 slice) ----
    v16h adjA;
    {
      V16H u;
      const _Float16* ap = &adjBuf[wr][m][0];
      u.q[0] = *(const uint4*)(ap + kb);
      u.q[1] = *(const uint4*)(ap + kb + 16);
      adjA = u.v;
    }
    #pragma unroll
    for (int ct = 0; ct < 8; ++ct) {
      const _Float16* bp = &skeyT[wc * 128 + ct * 16 + m][0];
      V16H u;
      u.q[0] = *(const uint4*)(bp + kb);
      u.q[1] = *(const uint4*)(bp + kb + 16);
      acc[ct] = wmma_f16(adjA, u.v, acc[ct]);
    }
  }

  // store y in f16 (feeds the projection GEMM only)
  #pragma unroll
  for (int ct = 0; ct < 8; ++ct) {
    #pragma unroll
    for (int r = 0; r < 8; ++r) {
      size_t row = batchOff + rowbase + r + rhalf;
      yh[row * D_ + wc * 128 + ct * 16 + m] = (_Float16)acc[ct][r];
    }
  }
}

// ---------------------------------------------------------------------------
// Kernel 3: out = elu(y @ W^T + b) + x      (WMMA GEMM + fused epilogue)
// ---------------------------------------------------------------------------
__launch_bounds__(256)
__global__ void proj_kernel(const _Float16* __restrict__ yh,
                            const _Float16* __restrict__ wh,
                            const float* __restrict__ bias,
                            const float* __restrict__ x,
                            float* __restrict__ out) {
  const int t    = threadIdx.x;
  const int wave = t >> 5, lane = t & 31;
  const int wr   = wave >> 1, wc = wave & 1;
  const int m    = lane & 15;
  const int kb   = (lane >> 4) * 8;
  const int rhalf = (lane >> 4) * 8;
  const size_t rowbase = (size_t)blockIdx.x * 64 + wr * 16;
  const int colbase = wc * 128;

  // hint next row-block of y toward the caches (global_prefetch_b8)
  __builtin_prefetch(yh + (rowbase + 64) * (size_t)D_ + t * 8, 0, 0);

  v8f zero = {};
  v8f acc[8];
  #pragma unroll
  for (int i = 0; i < 8; ++i) acc[i] = zero;

  const _Float16* ap = yh + (rowbase + m) * (size_t)D_;
  #pragma unroll
  for (int s = 0; s < 8; ++s) {
    V16H ua;
    ua.q[0] = *(const uint4*)(ap + s * 32 + kb);
    ua.q[1] = *(const uint4*)(ap + s * 32 + kb + 16);
    #pragma unroll
    for (int ct = 0; ct < 8; ++ct) {
      const _Float16* bp = wh + (size_t)(colbase + ct * 16 + m) * D_;
      V16H ub;
      ub.q[0] = *(const uint4*)(bp + s * 32 + kb);
      ub.q[1] = *(const uint4*)(bp + s * 32 + kb + 16);
      acc[ct] = wmma_f16(ua.v, ub.v, acc[ct]);
    }
  }

  #pragma unroll
  for (int ct = 0; ct < 8; ++ct) {
    int col = colbase + ct * 16 + m;
    float bv = bias[col];
    #pragma unroll
    for (int r = 0; r < 8; ++r) {
      size_t row = rowbase + r + rhalf;
      float v = acc[ct][r] + bv;
      v = v > 0.0f ? v : (__expf(v) - 1.0f);           // ELU (alpha=1)
      out[row * D_ + col] = v + x[row * D_ + col];     // residual
    }
  }
}

// ---------------------------------------------------------------------------
// Kernel 4: in-place LayerNorm over D=256, one block per row
// ---------------------------------------------------------------------------
__launch_bounds__(256)
__global__ void ln_kernel(float* __restrict__ out,
                          const float* __restrict__ g,
                          const float* __restrict__ be) {
  __shared__ float red[2][8];
  const int row = blockIdx.x, t = threadIdx.x;
  const int wave = t >> 5, lane = t & 31;
  float v = out[(size_t)row * D_ + t];
  float s = v, s2 = v * v;
  #pragma unroll
  for (int off = 16; off > 0; off >>= 1) {
    s  += __shfl_down(s, off);
    s2 += __shfl_down(s2, off);
  }
  if (lane == 0) { red[0][wave] = s; red[1][wave] = s2; }
  __syncthreads();
  if (wave == 0) {
    float a  = (lane < 8) ? red[0][lane] : 0.0f;
    float b2 = (lane < 8) ? red[1][lane] : 0.0f;
    #pragma unroll
    for (int off = 4; off > 0; off >>= 1) {
      a  += __shfl_down(a, off);
      b2 += __shfl_down(b2, off);
    }
    if (lane == 0) { red[0][0] = a; red[1][0] = b2; }
  }
  __syncthreads();
  float mean = red[0][0] * (1.0f / D_);
  float var  = red[1][0] * (1.0f / D_) - mean * mean;
  float rstd = rsqrtf(var + 1e-5f);
  out[(size_t)row * D_ + t] = (v - mean) * rstd * g[t] + be[t];
}

// ---------------------------------------------------------------------------
extern "C" void kernel_launch(void* const* d_in, const int* in_sizes, int n_in,
                              void* d_out, int out_size, void* d_ws, size_t ws_size,
                              hipStream_t stream) {
  (void)in_sizes; (void)n_in; (void)out_size; (void)ws_size;
  const float* x    = (const float*)d_in[0];
  const float* ls   = (const float*)d_in[1];
  const float* W    = (const float*)d_in[2];
  const float* bias = (const float*)d_in[3];
  const float* gam  = (const float*)d_in[4];
  const float* bet  = (const float*)d_in[5];
  char* ws = (char*)d_ws;

  const size_t XH = (size_t)B_ * N_ * D_ * sizeof(_Float16); // 8 MB
  const size_t WH = (size_t)D_ * D_ * sizeof(_Float16);      // 128 KB
  const size_t SQ = (size_t)B_ * N_ * sizeof(float);         // 64 KB
  _Float16* xh = (_Float16*)(ws);
  _Float16* wh = (_Float16*)(ws + XH);
  float*    sq = (float*)(ws + XH + WH);
  _Float16* yh = (_Float16*)(ws + XH + WH + SQ);             // 8 MB
  float* out = (float*)d_out;

  prep_x_kernel<<<(B_ * N_) / 8, 256, 0, stream>>>(x, xh, sq);
  prep_w_kernel<<<(D_ * D_) / 1024, 256, 0, stream>>>(W, wh);
  adj_agg_kernel<<<B_ * (N_ / 64), 256, 0, stream>>>(xh, sq, ls, yh);
  proj_kernel<<<(B_ * N_) / 64, 256, 0, stream>>>(yh, wh, bias, x, out);
  ln_kernel<<<B_ * N_, 256, 0, stream>>>(out, gam, bet);
}
